// VectorQuantizer_41257455845806
// MI455X (gfx1250) — matrix operand order repaired and checked by reading, not compiled, not run on hardware
//
#include <hip/hip_runtime.h>
#include <hip/hip_bf16.h>
#include <math.h>

// ---------------------------------------------------------------------------
// VQ-VAE vector quantizer for MI455X (gfx1250).
// Distance GEMM runs as split-F16 WMMA (x=xh+xl, e*2^10=eh+el, 3 products)
// giving FP32-accurate argmin at F16 matrix throughput.
// Sizes fixed by the reference: N=32768 rows, D=256, K=1024 codes.
// ---------------------------------------------------------------------------

typedef __attribute__((ext_vector_type(16))) _Float16 v16h;
typedef __attribute__((ext_vector_type(8)))  _Float16 v8h;
typedef __attribute__((ext_vector_type(4)))  _Float16 v4h;
typedef __attribute__((ext_vector_type(8)))  float    v8f;

#define NROWS 32768
#define DDIM  256
#define KCODE 1024
#define THW   2048                  // T*H*W = 8*16*16
#define Q_ELEMS (16 * 256 * 2048)   // 8388608
#define ENC_ELEMS ((size_t)NROWS * KCODE)
#define ESCALE 1024.0f              // 2^10: keeps el out of f16 denormals

union V16 { v16h v; v8h h[2]; };

// D = A(16x32 f16) * B(32x16 f16) + C(16x16 f32)  -- codegen-confirmed
__device__ __forceinline__ v8f wmma16(v16h a, v16h b, v8f c) {
  return __builtin_amdgcn_wmma_f32_16x16x32_f16(
      false, a, false, b, (short)0, c, false, false);
}

// ---------------------------------------------------------------------------
// Kernel 1: zero the one-hot encodings region of d_out and the stat scratch.
// Must run every call (counts are accumulated with atomics).
// ---------------------------------------------------------------------------
__global__ __launch_bounds__(256) void vq_zero(float* __restrict__ enc,
                                               int* __restrict__ counts) {
  size_t tid = (size_t)blockIdx.x * blockDim.x + threadIdx.x;
  size_t stride = (size_t)gridDim.x * blockDim.x;
  float2* p = (float2*)enc;                 // enc offset is 8B-aligned
  size_t n2 = ENC_ELEMS / 2;
  for (size_t i = tid; i < n2; i += stride) p[i] = make_float2(0.0f, 0.0f);
  if (tid < KCODE) counts[tid] = 0;
}

// ---------------------------------------------------------------------------
// Kernel 2: split codebook: es = e*2^10; eh = f16(es); el = f16(es - eh).
// Also e_norm_s[k] = 2^10 * sum_d e[k][d]^2 (exact power-of-2 scaling).
// One wave per code.
// ---------------------------------------------------------------------------
__global__ __launch_bounds__(256) void vq_prep(const float* __restrict__ emb,
                                               _Float16* __restrict__ eh,
                                               _Float16* __restrict__ el,
                                               float* __restrict__ enorm_s) {
  int wave = threadIdx.x >> 5;
  int lane = threadIdx.x & 31;
  int code = blockIdx.x * 8 + wave;         // grid = 128 -> 1024 codes
  const float* e = emb + (size_t)code * DDIM;
  float sum = 0.0f;
  for (int i = lane; i < DDIM; i += 32) {
    float v = e[i];
    sum += v * v;
    float vs = v * ESCALE;
    _Float16 h = (_Float16)vs;
    _Float16 l = (_Float16)(vs - (float)h);
    eh[(size_t)code * DDIM + i] = h;
    el[(size_t)code * DDIM + i] = l;
  }
  #pragma unroll
  for (int m = 16; m > 0; m >>= 1) sum += __shfl_xor(sum, m, 32);
  if (lane == 0) enorm_s[code] = sum * ESCALE;
}

// ---------------------------------------------------------------------------
// Kernel 3: fused GEMM (x . e^T) + argmin over codes, split-F16 WMMA.
// One wave per 16-row block (4 waves/block); 4 code-tiles per sweep of D.
// dist_scaled(row,code) = 2^10*||e||^2 - 2 * <x, 2^10*e>
//                       (||x||^2 dropped: constant per row; scale > 0)
// x rows are split to f16 hi/lo ONCE into LDS (xor-swizzled 8-half blocks,
// bank-conflict-free), then re-read as A-fragments via 16B ds loads.
// ---------------------------------------------------------------------------
__global__ __launch_bounds__(128) void vq_argmin_wmma(
    const float* __restrict__ in,         // [16, 256, 2048] (B, C, THW)
    const _Float16* __restrict__ eh,      // [1024, 256]
    const _Float16* __restrict__ el,      // [1024, 256]
    const float* __restrict__ enorm_s,    // [1024]
    int* __restrict__ enc_idx) {          // [32768]
  __shared__ __align__(16) _Float16 lds_xh[4][16][256];   // 32KB
  __shared__ __align__(16) _Float16 lds_xl[4][16][256];   // 32KB
  const int wave = threadIdx.x >> 5;
  const int lane = threadIdx.x & 31;
  const int half = lane >> 4;
  const int col  = lane & 15;
  const int rowbase = (blockIdx.x * 4 + wave) * 16;
  const int b = rowbase / THW;
  const int s = rowbase % THW;            // 16-aligned: block = one W-line
  const float* xbase = in + (size_t)b * DDIM * THW + s;

  // ---- Stage: split this wave's 16 x-rows into LDS (hi/lo f16) -----------
  // Element (row,d) stored at half-offset  ((d>>3)^row)*8 + (d&7)  (swizzle).
  {
    const int row = col;                   // lanes 0-15 / 16-31 share rows
    for (int i = 0; i < 32; ++i) {
      const int q  = i * 2 + half;         // d-quad 0..63
      const int d0 = q * 4;
      float v0 = xbase[(size_t)(d0 + 0) * THW + row];
      float v1 = xbase[(size_t)(d0 + 1) * THW + row];
      float v2 = xbase[(size_t)(d0 + 2) * THW + row];
      float v3 = xbase[(size_t)(d0 + 3) * THW + row];
      v4h hq, lq;
      hq[0] = (_Float16)v0; lq[0] = (_Float16)(v0 - (float)hq[0]);
      hq[1] = (_Float16)v1; lq[1] = (_Float16)(v1 - (float)hq[1]);
      hq[2] = (_Float16)v2; lq[2] = (_Float16)(v2 - (float)hq[2]);
      hq[3] = (_Float16)v3; lq[3] = (_Float16)(v3 - (float)hq[3]);
      const int off = (((d0 >> 3) ^ row) * 8) + (d0 & 7);   // 8B-aligned
      *(v4h*)(&lds_xh[wave][row][off]) = hq;
      *(v4h*)(&lds_xl[wave][row][off]) = lq;
    }
  }
  __syncthreads();

  float minv[8];
  int   mini[8];
  #pragma unroll
  for (int j = 0; j < 8; ++j) { minv[j] = 3.402823e38f; mini[j] = 0; }

  const _Float16* xr_h = &lds_xh[wave][col][0];
  const _Float16* xr_l = &lds_xl[wave][col][0];

  for (int kg = 0; kg < 16; ++kg) {        // 16 groups x 4 tiles = 64 tiles
    v8f acc0 = {}, acc1 = {}, acc2 = {}, acc3 = {};
    const size_t crow = (size_t)((kg * 4) * 16 + col) * DDIM;
    const _Float16* eh0 = eh + crow;       // tile t at +t*16*DDIM
    const _Float16* el0 = el + crow;
    if (kg + 1 < 16) {                     // pull next code group toward L2/L0
      __builtin_prefetch(eh0 + 64 * DDIM, 0, 0);   // -> global_prefetch_b8
      __builtin_prefetch(el0 + 64 * DDIM, 0, 0);
    }
    #pragma unroll
    for (int c = 0; c < 8; ++c) {          // 8 chunks of K=32 over D=256
      // A-fragments: contiguous k order (not ISA table order) -> lane holds K = half*16+{0..15}
      // of row `col` (matches B).
      const int g0 = c * 32 + half * 16;
      const int g1 = g0 + 8;
      const int o0 = ((g0 >> 3) ^ col) * 8;
      const int o1 = ((g1 >> 3) ^ col) * 8;
      V16 ah, al;
      ah.h[0] = *(const v8h*)(xr_h + o0);
      ah.h[1] = *(const v8h*)(xr_h + o1);
      al.h[0] = *(const v8h*)(xr_l + o0);
      al.h[1] = *(const v8h*)(xr_l + o1);
      // B-fragments: contiguous k order (same as A) -> lane holds K = half*16+{0..15} of
      // codebook row (tile_base + col); contiguous 32B loads.
      const int boff = c * 32 + half * 16;
      v16h bh0 = *(const v16h*)(eh0 + 0 * 16 * DDIM + boff);
      v16h bl0 = *(const v16h*)(el0 + 0 * 16 * DDIM + boff);
      v16h bh1 = *(const v16h*)(eh0 + 1 * 16 * DDIM + boff);
      v16h bl1 = *(const v16h*)(el0 + 1 * 16 * DDIM + boff);
      v16h bh2 = *(const v16h*)(eh0 + 2 * 16 * DDIM + boff);
      v16h bl2 = *(const v16h*)(el0 + 2 * 16 * DDIM + boff);
      v16h bh3 = *(const v16h*)(eh0 + 3 * 16 * DDIM + boff);
      v16h bl3 = *(const v16h*)(el0 + 3 * 16 * DDIM + boff);
      // Split product: xh*eh + xl*eh + xh*el (xl*el ~2^-22, dropped)
      acc0 = wmma16(ah.v, bh0, acc0);
      acc1 = wmma16(ah.v, bh1, acc1);
      acc2 = wmma16(ah.v, bh2, acc2);
      acc3 = wmma16(ah.v, bh3, acc3);
      acc0 = wmma16(al.v, bh0, acc0);
      acc1 = wmma16(al.v, bh1, acc1);
      acc2 = wmma16(al.v, bh2, acc2);
      acc3 = wmma16(al.v, bh3, acc3);
      acc0 = wmma16(ah.v, bl0, acc0);
      acc1 = wmma16(ah.v, bl1, acc1);
      acc2 = wmma16(ah.v, bl2, acc2);
      acc3 = wmma16(ah.v, bl3, acc3);
    }
    // Epilogue: branchless running min. C layout: VGPR j holds row (j+8*half)
    // for column `col` of the tile.
    #pragma unroll
    for (int t = 0; t < 4; ++t) {
      const int code = (kg * 4 + t) * 16 + col;
      const float en = enorm_s[code];
      const v8f& acc = (t == 0) ? acc0 : (t == 1) ? acc1 : (t == 2) ? acc2 : acc3;
      #pragma unroll
      for (int j = 0; j < 8; ++j) {
        float dist = en - 2.0f * acc[j];
        bool better = dist < minv[j];      // strict < : lowest code wins ties
        mini[j] = better ? code : mini[j];
        minv[j] = better ? dist : minv[j];
      }
    }
  }

  // Cross-lane min+index reduce over the 16 columns of each half
  // (xor masks < 16 never cross the half boundary).
  #pragma unroll
  for (int j = 0; j < 8; ++j) {
    float v = minv[j];
    int   ix = mini[j];
    #pragma unroll
    for (int m = 1; m < 16; m <<= 1) {
      float ov = __shfl_xor(v, m, 32);
      int   oi = __shfl_xor(ix, m, 32);
      bool take = (ov < v) || (ov == v && oi < ix);   // argmin: lowest index
      v  = take ? ov : v;
      ix = take ? oi : ix;
    }
    if (col == j) enc_idx[rowbase + half * 8 + j] = ix;
  }
}

// ---------------------------------------------------------------------------
// Kernel 4: gather quantized rows (exact f32 codebook), write NCTHW output,
// one-hot scatter, code counts, deterministic squared-error partials.
// One thread per row; 128 blocks x 256 threads.
// ---------------------------------------------------------------------------
__global__ __launch_bounds__(256) void vq_gather(
    const float* __restrict__ in, const float* __restrict__ emb,
    const int* __restrict__ enc_idx, float* __restrict__ q_out,
    float* __restrict__ encodings, int* __restrict__ counts,
    float* __restrict__ partials) {
  const int n = blockIdx.x * 256 + threadIdx.x;
  const int b = n / THW;
  const int s = n % THW;
  const int idx = enc_idx[n];
  const float* e = emb + (size_t)idx * DDIM;
  const float* x = in + (size_t)b * DDIM * THW + s;
  float* q = q_out + (size_t)b * DDIM * THW + s;
  float sum = 0.0f;
  #pragma unroll 4
  for (int d = 0; d < DDIM; ++d) {
    float qv = e[d];
    float xv = x[(size_t)d * THW];
    q[(size_t)d * THW] = qv;               // coalesced across threads (s)
    float df = qv - xv;
    sum += df * df;
  }
  encodings[(size_t)n * KCODE + idx] = 1.0f;
  atomicAdd(&counts[idx], 1);

  __shared__ float red[256];
  red[threadIdx.x] = sum;
  __syncthreads();
  for (int off = 128; off > 0; off >>= 1) {
    if (threadIdx.x < off) red[threadIdx.x] += red[threadIdx.x + off];
    __syncthreads();
  }
  if (threadIdx.x == 0) partials[blockIdx.x] = red[0];
}

// ---------------------------------------------------------------------------
// Kernel 5: finalize loss and perplexity (deterministic reductions).
// ---------------------------------------------------------------------------
__global__ __launch_bounds__(256) void vq_finalize(
    const int* __restrict__ counts, const float* __restrict__ partials,
    float* __restrict__ loss_out, float* __restrict__ perp_out) {
  __shared__ float red[256];
  float ent = 0.0f;
  for (int k = threadIdx.x; k < KCODE; k += 256) {
    float p = (float)counts[k] * (1.0f / (float)NROWS);
    ent += p * logf(p + 1e-10f);
  }
  red[threadIdx.x] = ent;
  __syncthreads();
  for (int off = 128; off > 0; off >>= 1) {
    if (threadIdx.x < off) red[threadIdx.x] += red[threadIdx.x + off];
    __syncthreads();
  }
  if (threadIdx.x == 0) {
    float mse = 0.0f;
    for (int i = 0; i < 128; ++i) mse += partials[i];   // fixed order
    mse /= (float)((size_t)NROWS * DDIM);
    loss_out[0] = 1.25f * mse;               // mse + 0.25*mse
    perp_out[0] = expf(-red[0]);
  }
}

// ---------------------------------------------------------------------------
// Host launcher. Output layout (flat float32, reference return order):
//   [ loss(1) | quantized(8388608) | perplexity(1) | encodings(33554432) ]
// ---------------------------------------------------------------------------
extern "C" void kernel_launch(void* const* d_in, const int* in_sizes, int n_in,
                              void* d_out, int out_size, void* d_ws, size_t ws_size,
                              hipStream_t stream) {
  const float* in  = (const float*)d_in[0];   // [16,256,8,16,16]
  const float* emb = (const float*)d_in[1];   // [1024,256]
  float* out   = (float*)d_out;
  float* q_out = out + 1;
  float* perp  = out + 1 + Q_ELEMS;
  float* enc   = out + 2 + Q_ELEMS;           // offset 8388610 (8B-aligned)

  // Workspace layout (~1.2 MB)
  float*    enorm_s  = (float*)d_ws;                 // 1024 f32
  int*      enc_idx  = (int*)(enorm_s + KCODE);      // 32768 i32
  int*      counts   = (int*)(enc_idx + NROWS);      // 1024 i32
  float*    partials = (float*)(counts + KCODE);     // 128 f32
  _Float16* eh       = (_Float16*)(partials + 128);  // 1024*256 f16
  _Float16* el       = eh + (size_t)KCODE * DDIM;    // 1024*256 f16

  vq_zero       <<<2048, 256, 0, stream>>>(enc, counts);
  vq_prep       <<<128,  256, 0, stream>>>(emb, eh, el, enorm_s);
  vq_argmin_wmma<<<512,  128, 0, stream>>>(in, eh, el, enorm_s, enc_idx);
  vq_gather     <<<128,  256, 0, stream>>>(in, emb, enc_idx, q_out, enc,
                                           counts, partials);
  vq_finalize   <<<1,    256, 0, stream>>>(counts, partials, out, perp);
}
